// ShiftedWindowMultiHeadAttention_48868137893997
// MI455X (gfx1250) — matrix operand order repaired and checked by reading, not compiled, not run on hardware
//
#include <hip/hip_runtime.h>

typedef __attribute__((ext_vector_type(16))) __bf16        v16bf;
typedef __attribute__((ext_vector_type(8)))  float         v8f;
typedef __attribute__((ext_vector_type(8)))  unsigned int  v8u;

union FragBF { v16bf bf; v8u u; };

#define IMG   224
#define CH    96
#define WSZ   7
#define SHF   3
#define NTOK  49      // WSZ*WSZ
#define NP    64      // padded tokens (4 WMMA tiles)
#define HEADS 3
#define HD    32
#define WIN_PER_IMG 1024   // 32*32
#define TOT_WIN     8192   // 8 * 1024
#define QKV_C 288

// region label for shifted-window mask: [0,217) -> 0, [217,221) -> 1, [221,224) -> 2
__device__ __forceinline__ int reg3(int p) { return (p < IMG - WSZ) ? 0 : ((p < IMG - SHF) ? 1 : 2); }
// paired K indices (elements 2t, 2t+1 of a fragment hold K and K+1) — ISA 7.12.2
__device__ __forceinline__ int a_k2(int t, int hf) { return 2 * t + 8 * hf + ((t >= 4) ? 8 : 0); }
__device__ __forceinline__ int b_k2(int t, int hf) { return 2 * t + 8 * hf + ((t >= 4) ? 8 : 0); }

// ---------------- Kernel 1: shifted gather + QKV GEMM (bf16 WMMA) ----------------
__global__ __launch_bounds__(256) void k_qkv(const float* __restrict__ x,
                                             const float* __restrict__ qkv_w,
                                             const float* __restrict__ qkv_b,
                                             void* __restrict__ qkv_ws_raw)
{
    __bf16* __restrict__ qkvws = (__bf16*)qkv_ws_raw;
    __shared__ __bf16 Xs[NP][CH + 8];
    __shared__ __bf16 WsT[QKV_C][CH + 8];   // transposed bf16 weights: WsT[col][k]

    const int win  = blockIdx.x;
    const int b    = win >> 10;
    const int wloc = win & (WIN_PER_IMG - 1);
    const int wy   = wloc >> 5;
    const int wx   = wloc & 31;
    const int tid  = threadIdx.x;

    // stage shifted window tile (f32 -> bf16); rows >= 49 clamp to row 48 (harmless pad)
    for (int e = tid; e < NP * (CH / 4); e += 256) {
        int i = e / (CH / 4), c4 = e - i * (CH / 4);
        int ic = (i < NTOK) ? i : (NTOK - 1);
        int yi = ic / WSZ, xi = ic - yi * WSZ;
        int py = wy * WSZ + yi + SHF; if (py >= IMG) py -= IMG;
        int px = wx * WSZ + xi + SHF; if (px >= IMG) px -= IMG;
        float4 v = *(const float4*)&x[((size_t)(b * IMG + py) * IMG + px) * CH + 4 * c4];
        Xs[i][4 * c4 + 0] = (__bf16)v.x;
        Xs[i][4 * c4 + 1] = (__bf16)v.y;
        Xs[i][4 * c4 + 2] = (__bf16)v.z;
        Xs[i][4 * c4 + 3] = (__bf16)v.w;
    }
    // stage qkv_w transposed (96x288 f32 -> WsT[288][96] bf16)
    for (int e = tid; e < CH * (QKV_C / 2); e += 256) {
        int r = e / (QKV_C / 2), c2 = e - r * (QKV_C / 2);
        float2 v = *(const float2*)&qkv_w[(size_t)r * QKV_C + 2 * c2];
        WsT[2 * c2 + 0][r] = (__bf16)v.x;
        WsT[2 * c2 + 1][r] = (__bf16)v.y;
    }
    __syncthreads();

    const int wave = tid >> 5, lane = tid & 31, hf = lane >> 4, ln = lane & 15;
    const float qscale = 0.17677669529663687f;  // HD^-0.5

    // 4 M-tiles x 18 N-tiles, round-robin over 8 waves
    for (int tile = wave; tile < 72; tile += 8) {
        const int mt = tile & 3, nt = tile >> 2;
        const int arow = mt * 16 + ln;
        const int col  = nt * 16 + ln;
        v8f acc = {};
#pragma unroll
        for (int ks = 0; ks < 3; ++ks) {
            FragBF a, bb;
#pragma unroll
            for (int t = 0; t < 8; ++t) {
                a.u[t]  = *(const unsigned int*)&Xs[arow][ks * 32 + a_k2(t, hf)];
                bb.u[t] = *(const unsigned int*)&WsT[col][ks * 32 + b_k2(t, hf)];
            }
            acc = __builtin_amdgcn_wmma_f32_16x16x32_bf16(false, a.bf, false, bb.bf,
                                                          (short)0, acc, false, false);
        }
        const float bias = qkv_b[col];
        const float scl  = (col < CH) ? qscale : 1.f;  // scale only Q columns
#pragma unroll
        for (int r = 0; r < 8; ++r) {
            int i = mt * 16 + r + 8 * hf;
            qkvws[((size_t)win * NP + i) * QKV_C + col] = (__bf16)((acc[r] + bias) * scl);
        }
    }
}

// -------- Kernel 2: per-(window, head) attention: QK^T + bias + mask + softmax + PV --------
__global__ __launch_bounds__(128) void k_attn(const void* __restrict__ qkv_ws_raw,
                                              const float* __restrict__ rel_bias,
                                              void* __restrict__ attn_ws_raw)
{
    const unsigned int* __restrict__ qw32 = (const unsigned int*)qkv_ws_raw;  // paired bf16
    __bf16* __restrict__ attnws = (__bf16*)attn_ws_raw;

    __shared__ __bf16 Qs[NP][HD + 4];
    __shared__ __bf16 Ks[NP][HD + 4];
    __shared__ __bf16 Vt[HD][NP + 4];      // V transposed: Vt[d][token]
    __shared__ float  Sc[NP][NP + 4];
    __shared__ __bf16 Ps[NP][NP + 4];

    const int wh   = blockIdx.x;
    const int win  = wh / HEADS;
    const int h    = wh - win * HEADS;
    const int wloc = win & (WIN_PER_IMG - 1);
    const int wy   = wloc >> 5;
    const int wx   = wloc & 31;
    const int tid  = threadIdx.x;

    // stage Q/K (row-major) and V (transposed); padded ws rows are finite garbage
    for (int e = tid; e < NP * (HD / 2); e += 128) {
        int i = e >> 4, d2 = e & 15;
        size_t base = (((size_t)win * NP + i) * QKV_C + h * HD + 2 * d2) >> 1;
        unsigned int q = qw32[base];
        unsigned int k = qw32[base + CH / 2];
        unsigned int v = qw32[base + CH];     // 2*CH bf16 -> CH uints
        *(unsigned int*)&Qs[i][2 * d2] = q;
        *(unsigned int*)&Ks[i][2 * d2] = k;
        union { unsigned int u; __bf16 b[2]; } vv; vv.u = v;
        Vt[2 * d2 + 0][i] = vv.b[0];
        Vt[2 * d2 + 1][i] = vv.b[1];
    }
    __syncthreads();

    const int mt = tid >> 5, lane = tid & 31, hf = lane >> 4, ln = lane & 15;

    // scores: wave mt owns rows [mt*16, mt*16+16)
#pragma unroll
    for (int nt = 0; nt < 4; ++nt) {
        FragBF a, bb;
#pragma unroll
        for (int t = 0; t < 8; ++t) {
            a.u[t]  = *(const unsigned int*)&Qs[mt * 16 + ln][a_k2(t, hf)];
            bb.u[t] = *(const unsigned int*)&Ks[nt * 16 + ln][b_k2(t, hf)];
        }
        v8f acc = {};
        acc = __builtin_amdgcn_wmma_f32_16x16x32_bf16(false, a.bf, false, bb.bf,
                                                      (short)0, acc, false, false);
        const int j = nt * 16 + ln;
        const int yj = j / WSZ, xj = j - yj * WSZ;
#pragma unroll
        for (int r = 0; r < 8; ++r) {
            int i = mt * 16 + r + 8 * hf;
            float val = -1e30f;
            if (i < NTOK && j < NTOK) {
                int yi = i / WSZ, xi = i - yi * WSZ;
                int ridx = (yi - yj + WSZ - 1) * (2 * WSZ - 1) + (xi - xj + WSZ - 1);
                val = acc[r] + rel_bias[ridx * HEADS + h];
                int li = reg3(wy * WSZ + yi) * 3 + reg3(wx * WSZ + xi);
                int lj = reg3(wy * WSZ + yj) * 3 + reg3(wx * WSZ + xj);
                if (li != lj) val -= 100.f;
            }
            Sc[i][j] = val;
        }
    }
    __syncthreads();

    // row softmax: lanes 0..15 of wave mt each own one row
    if (lane < 16) {
        const int i = mt * 16 + lane;
        float m = -1e30f;
        for (int j = 0; j < NTOK; ++j) m = fmaxf(m, Sc[i][j]);
        float s = 0.f;
        for (int j = 0; j < NTOK; ++j) s += __expf(Sc[i][j] - m);
        const float inv = 1.f / s;
        for (int j = 0; j < NP; ++j)
            Ps[i][j] = (__bf16)((j < NTOK) ? __expf(Sc[i][j] - m) * inv : 0.f);
    }
    __syncthreads();

    // out = P @ V : 2 d-tiles, K = 64 tokens -> 2 k-steps
#pragma unroll
    for (int ntd = 0; ntd < 2; ++ntd) {
        v8f acc = {};
#pragma unroll
        for (int ks = 0; ks < 2; ++ks) {
            FragBF a, bb;
#pragma unroll
            for (int t = 0; t < 8; ++t) {
                a.u[t]  = *(const unsigned int*)&Ps[mt * 16 + ln][ks * 32 + a_k2(t, hf)];
                bb.u[t] = *(const unsigned int*)&Vt[ntd * 16 + ln][ks * 32 + b_k2(t, hf)];
            }
            acc = __builtin_amdgcn_wmma_f32_16x16x32_bf16(false, a.bf, false, bb.bf,
                                                          (short)0, acc, false, false);
        }
        const int d = ntd * 16 + ln;
#pragma unroll
        for (int r = 0; r < 8; ++r) {
            int i = mt * 16 + r + 8 * hf;
            attnws[((size_t)win * NP + i) * CH + h * HD + d] = (__bf16)acc[r];
        }
    }
}

// ---------------- Kernel 3: output projection + inverse roll scatter + residual ----------------
__global__ __launch_bounds__(256) void k_proj(const void* __restrict__ attn_ws_raw,
                                              const float* __restrict__ proj_w,
                                              const float* __restrict__ proj_b,
                                              const float* __restrict__ x,
                                              float* __restrict__ out)
{
    const unsigned int* __restrict__ aw32 = (const unsigned int*)attn_ws_raw;  // paired bf16
    __shared__ __bf16 As[NP][CH + 8];
    __shared__ __bf16 WpT[CH][CH + 8];    // transposed bf16 proj weights: WpT[col][k]

    const int win  = blockIdx.x;
    const int b    = win >> 10;
    const int wloc = win & (WIN_PER_IMG - 1);
    const int wy   = wloc >> 5;
    const int wx   = wloc & 31;
    const int tid  = threadIdx.x;

    for (int e = tid; e < NP * (CH / 2); e += 256) {
        int i = e / (CH / 2), c2 = e - i * (CH / 2);
        unsigned int v = aw32[(((size_t)win * NP + i) * CH + 2 * c2) >> 1];
        *(unsigned int*)&As[i][2 * c2] = v;
    }
    for (int e = tid; e < CH * (CH / 2); e += 256) {
        int r = e / (CH / 2), c2 = e - r * (CH / 2);
        float2 v = *(const float2*)&proj_w[(size_t)r * CH + 2 * c2];
        WpT[2 * c2 + 0][r] = (__bf16)v.x;
        WpT[2 * c2 + 1][r] = (__bf16)v.y;
    }
    __syncthreads();

    const int wave = tid >> 5, lane = tid & 31, hf = lane >> 4, ln = lane & 15;

    // 4 M-tiles x 6 N-tiles over 8 waves
    for (int tile = wave; tile < 24; tile += 8) {
        const int mt = tile & 3, nt = tile >> 2;
        const int arow = mt * 16 + ln;
        const int col  = nt * 16 + ln;
        v8f acc = {};
#pragma unroll
        for (int ks = 0; ks < 3; ++ks) {
            FragBF a, bb;
#pragma unroll
            for (int t = 0; t < 8; ++t) {
                a.u[t]  = *(const unsigned int*)&As[arow][ks * 32 + a_k2(t, hf)];
                bb.u[t] = *(const unsigned int*)&WpT[col][ks * 32 + b_k2(t, hf)];
            }
            acc = __builtin_amdgcn_wmma_f32_16x16x32_bf16(false, a.bf, false, bb.bf,
                                                          (short)0, acc, false, false);
        }
        const float pb = proj_b[col];
#pragma unroll
        for (int r = 0; r < 8; ++r) {
            int i = mt * 16 + r + 8 * hf;
            if (i < NTOK) {   // real output scatter: guard required
                int yi = i / WSZ, xi = i - yi * WSZ;
                int py = wy * WSZ + yi + SHF; if (py >= IMG) py -= IMG;
                int px = wx * WSZ + xi + SHF; if (px >= IMG) px -= IMG;
                size_t idx = ((size_t)(b * IMG + py) * IMG + px) * CH + col;
                out[idx] = x[idx] + acc[r] + pb;   // fused residual
            }
        }
    }
}

extern "C" void kernel_launch(void* const* d_in, const int* in_sizes, int n_in,
                              void* d_out, int out_size, void* d_ws, size_t ws_size,
                              hipStream_t stream) {
    (void)in_sizes; (void)n_in; (void)out_size; (void)ws_size;
    const float* x      = (const float*)d_in[0];
    const float* qkv_w  = (const float*)d_in[1];
    const float* qkv_b  = (const float*)d_in[2];
    const float* proj_w = (const float*)d_in[3];
    const float* proj_b = (const float*)d_in[4];
    const float* relb   = (const float*)d_in[5];
    float* out = (float*)d_out;

    // workspace layout (bf16, NP-padded rows):
    //   qkv  : 8192 * 64 * 288 elems = 301,989,888 bytes
    //   attn : 8192 * 64 *  96 elems = 100,663,296 bytes
    void* qkvws  = d_ws;
    void* attnws = (void*)((char*)d_ws + (size_t)TOT_WIN * NP * QKV_C * 2);

    k_qkv <<<TOT_WIN,          256, 0, stream>>>(x, qkv_w, qkv_b, qkvws);
    k_attn<<<TOT_WIN * HEADS,  128, 0, stream>>>(qkvws, relb, attnws);
    k_proj<<<TOT_WIN,          256, 0, stream>>>(attnws, proj_w, proj_b, x, out);
}
